// HamiltonianFlowModel_33148557590757
// MI455X (gfx1250) — matrix operand order repaired and checked by reading, not compiled, not run on hardware
//
#include <hip/hip_runtime.h>

#define BATCH   131072
#define DIM     64
#define QD      32
#define HF      128
#define HH      256
#define NLAYERS 6
#define DTS     0.05f

typedef __attribute__((ext_vector_type(16))) _Float16 v16h;
typedef __attribute__((ext_vector_type(8)))  _Float16 v8h;
typedef __attribute__((ext_vector_type(8)))  float    v8f;

// LDS strides (elements), +8-half padding => row stride = 4 banks mod 64
#define SLD 72    // f32 state  [16 samples][72]
#define HLD 264   // f16 hidden [16 samples][264]
#define MLD 72    // f16 M      [64][72]
#define GLD 72    // f16 gradH  [16 samples][72]

// byte offsets into dynamic LDS
#define SBUF_OFF   0         // 4 waves * 16*72*4  = 18432
#define H1_OFF     18432     // 4 waves * 16*264*2 = 33792
#define H2_OFF     52224
#define H3_OFF     86016
#define M_OFF      119808    // 64*72*2 = 9216
#define ARENA_OFF  129024    // weight staging arena (172032 used) + const tail
#define CST_OFF    (ARENA_OFF + 172032)  // 13952 bytes of biases/w3/Bc (f32)
#define SMEM_BYTES 321536    // <= 320KB WGP LDS

// ---- pre-converted/padded/transposed f16 weight images in device-global memory ----
#define IMG_FLOW(L)  ((L)*18944)         // per layer: s1[128][40]@0, s2[32][136]@5120, t1@9472, t2@14592
#define IMG_KW1   113664                 // [256][72]
#define IMG_KW2   132096                 // [256][264]
#define IMG_KW1T  199680                 // [64][264]
#define IMG_KW2T  216576                 // [256][264]
#define IMG_VW1   284160                 // [256][40]
#define IMG_VW2   294400                 // [256][264]
#define IMG_VW1T  361984                 // [32][264]
#define IMG_VW2T  370432                 // [256][264]
#define IMG_TOTAL 438016
__device__ __align__(16) _Float16 g_img[IMG_TOTAL];

#if defined(__AMDGCN__) && __has_builtin(__builtin_amdgcn_tensor_load_to_lds) && __has_builtin(__builtin_amdgcn_s_wait_tensorcnt)
#define USE_TDM 1
#else
#define USE_TDM 0
#endif

__device__ __forceinline__ void memfence(){ asm volatile("" ::: "memory"); }
__device__ __forceinline__ float sigmoidf_(float x){ return 1.f/(1.f+__expf(-x)); }
__device__ __forceinline__ float softplusf_(float x){
  return fmaxf(x,0.f) + __logf(1.f + __expf(-fabsf(x)));
}
__device__ __forceinline__ float tanhf_(float x){
  float e = __expf(2.f*x);
  return 1.f - 2.f/(e + 1.f);
}

// ---- WMMA fragment builders (wave32, 16x16x32 f16), transposed formulation ----
__device__ __forceinline__ v16h a_frag_w(const _Float16* W, int ldw, int m0, int k0, int lane){
  int m    = m0 + (lane & 15);
  int base = k0 + ((lane >> 4) << 4);
  const v8h* p0 = (const v8h*)(W + m*ldw + base);
  const v8h* p1 = (const v8h*)(W + m*ldw + base + 8);
  v8h x = *p0, y = *p1;
  v16h a;
  #pragma unroll
  for (int j=0;j<8;++j){ a[j]=x[j]; a[j+8]=y[j]; }
  return a;
}
__device__ __forceinline__ v16h b_frag_x16(const _Float16* X, int ldx, int k0, int lane){
  int n  = lane & 15;
  int kb = k0 + ((lane >> 4) << 4);
  const v8h* p = (const v8h*)(X + n*ldx + kb);
  v8h x = p[0], y = p[1];
  v16h b;
  #pragma unroll
  for (int j=0;j<8;++j){ b[j]=x[j]; b[j+8]=y[j]; }
  return b;
}
__device__ __forceinline__ v16h b_frag_x32(const float* X, int ldx, int k0, int lane){
  int n  = lane & 15;
  int kb = k0 + ((lane >> 4) << 4);
  const float* p = X + n*ldx + kb;
  v16h b;
  #pragma unroll
  for (int j=0;j<16;++j) b[j] = (_Float16)p[j];
  return b;
}
__device__ __forceinline__ v8f wmma_f16(v16h a, v16h b, v8f c){
  return __builtin_amdgcn_wmma_f32_16x16x32_f16(false, a, false, b, (short)0, c, false, false);
}

// Y^T = act(W @ X^T + bias), stored row-major Y[sample][feat] with packed b128 stores.
template<int ACT, bool XF32, int KIN>
__device__ __forceinline__ void denseT(const _Float16* W, int ldw, int Nout,
    const void* X, int ldx, const float* bias,
    _Float16* Y, int ldy, _Float16* Ysig, const float* w3, int lane)
{
  const int lh = lane >> 4, lc = lane & 15;
  constexpr int NB = KIN / 32;
  constexpr bool PRE = (KIN <= 64);
  v16h bpre[PRE ? NB : 1];
  if (PRE){
    #pragma unroll
    for (int k = 0; k < NB; ++k){
      if (XF32) bpre[k] = b_frag_x32((const float*)X, ldx, 32*k, lane);
      else      bpre[k] = b_frag_x16((const _Float16*)X, ldx, 32*k, lane);
    }
  }
  #pragma unroll 2
  for (int m0 = 0; m0 < Nout; m0 += 16){
    v8f c = {};
    #pragma unroll
    for (int k = 0; k < NB; ++k){
      v16h a = a_frag_w(W, ldw, m0, 32*k, lane);
      v16h b;
      if (PRE)       b = bpre[k];
      else if (XF32) b = b_frag_x32((const float*)X, ldx, 32*k, lane);
      else           b = b_frag_x16((const _Float16*)X, ldx, 32*k, lane);
      c = wmma_f16(a, b, c);
    }
    const float* bp  = bias + m0 + 8*lh;
    const float* w3p = w3   + m0 + 8*lh;   // only deref for ACT==4
    v8h yh, sh;
    #pragma unroll
    for (int r = 0; r < 8; ++r){
      float x = c[r] + bp[r];
      float y;
      if      (ACT==0) y = fmaxf(x, 0.f);
      else if (ACT==1) y = tanhf_(x);
      else if (ACT==3){ y = softplusf_(x); sh[r] = (_Float16)sigmoidf_(x); }
      else if (ACT==4) y = w3p[r] * sigmoidf_(x);
      else             y = x;
      yh[r] = (_Float16)y;
    }
    *(v8h*)(Y + lc*ldy + m0 + 8*lh) = yh;
    if (ACT==3) *(v8h*)(Ysig + lc*ldy + m0 + 8*lh) = sh;
  }
}

// G1^T = (W2^T @ G2^T) * SIG
__device__ __forceinline__ void grad_hiddenT(const _Float16* G2, const _Float16* SIG,
    const _Float16* W2T, _Float16* G1, int lane){
  const int lh = lane >> 4, lc = lane & 15;
  #pragma unroll 2
  for (int m0 = 0; m0 < HH; m0 += 16){
    v8f c = {};
    #pragma unroll
    for (int k0 = 0; k0 < HH; k0 += 32)
      c = wmma_f16(a_frag_w(W2T, HLD, m0, k0, lane), b_frag_x16(G2, HLD, k0, lane), c);
    v8h sg = *(const v8h*)(SIG + lc*HLD + m0 + 8*lh);
    v8h g;
    #pragma unroll
    for (int r=0;r<8;++r) g[r] = (_Float16)(c[r] * (float)sg[r]);
    *(v8h*)(G1 + lc*HLD + m0 + 8*lh) = g;
  }
}

// fallback: pure f16 image copy global -> LDS, 16B granules
__device__ __forceinline__ void stage_cp(_Float16* dst, const _Float16* src, int halves, int tid){
  const v8h* s = (const v8h*)src;
  v8h* d = (v8h*)dst;
  int n = halves >> 3;
  for (int i = tid; i < n; i += 128) d[i] = s[i];
}
__device__ __forceinline__ void copy_f32(float* dst, const float* src, int n, int tid){
  for (int i = tid; i < n; i += 128) dst[i] = src[i];
}

#if USE_TDM
typedef __attribute__((ext_vector_type(4))) unsigned int u32x4;
typedef __attribute__((ext_vector_type(8))) int i32x8;
typedef __attribute__((ext_vector_type(4))) int i32x4;
// Issue one TDM load: d0 = elements per row (or total for 1D, d1==0), d1 = rows.
__device__ __forceinline__ void tdm_load(unsigned lds_byte, const void* gsrc,
                                         unsigned d0, unsigned d1){
  unsigned long long ga = (unsigned long long)(uintptr_t)gsrc;
  unsigned td1 = d1 ? d1 : 1u;
  u32x4 g0;
  g0[0] = 1u;                                   // count=1, user mode
  g0[1] = lds_byte;                             // lds_addr
  g0[2] = (unsigned)ga;                         // global_addr[31:0] -> bits 95:64
  g0[3] = (unsigned)(ga >> 32) | (2u << 30);    // global_addr[56:32], type=2
  i32x8 g1;
  g1[0] = (int)(1u << 16);                                      // data_size=1 (2B)
  g1[1] = (int)((d0 & 0xffffu) << 16);                          // tensor_dim0 lo
  g1[2] = (int)((d0 >> 16) | ((td1 & 0xffffu) << 16));          // tdim0 hi | tensor_dim1 lo
  g1[3] = (int)((td1 >> 16) | ((d0 & 0xffffu) << 16));          // tdim1 hi | tile_dim0
  g1[4] = (int)(d1 & 0xffffu);                                  // tile_dim1 (0 => 1D), tile_dim2=0
  g1[5] = (int)d0;                                              // tensor_dim0_stride[31:0]
  g1[6] = 0;
  g1[7] = 0;
  i32x4 z4 = {0,0,0,0};
  i32x8 z8 = {0,0,0,0,0,0,0,0};
  __builtin_amdgcn_tensor_load_to_lds(g0, g1, z4, z4, z8, 0);
}
#endif

// stage one weight image into LDS at byte offset lds_off
__device__ __forceinline__ void stage_img(char* smem, unsigned lds_off, const _Float16* src,
                                          unsigned d0, unsigned d1, int tid){
#if USE_TDM
  if (tid < 32){
    tdm_load(lds_off, src, d0, d1);
    __builtin_amdgcn_s_wait_tensorcnt(0);
  }
#else
  stage_cp((_Float16*)(smem + lds_off), src, (int)(d0 * (d1 ? d1 : 1u)), tid);
#endif
}

// ---------- prep: convert/pad/transpose all weights into g_img (f16) ----------
__global__ void __launch_bounds__(256)
prep_kernel(const float* __restrict__ sW1, const float* __restrict__ sW2,
            const float* __restrict__ tW1, const float* __restrict__ tW2,
            const float* __restrict__ kW1, const float* __restrict__ kW2,
            const float* __restrict__ vW1, const float* __restrict__ vW2)
{
  const int tidg = blockIdx.x*256 + threadIdx.x;
  const int nth  = gridDim.x*256;
  auto img = [&](int off, const float* src, int rows, int cols, int ld, bool tr){
    int tot = rows*cols;
    for (int i = tidg; i < tot; i += nth){
      int r = i / cols, c = i - r*cols;
      g_img[off + (tr ? c*ld + r : r*ld + c)] = (_Float16)src[i];
    }
  };
  for (int L = 0; L < NLAYERS; ++L){
    img(IMG_FLOW(L)        , sW1 + L*HF*QD, HF, QD, 40,  false);
    img(IMG_FLOW(L) + 5120 , sW2 + L*QD*HF, QD, HF, 136, false);
    img(IMG_FLOW(L) + 9472 , tW1 + L*HF*QD, HF, QD, 40,  false);
    img(IMG_FLOW(L) + 14592, tW2 + L*QD*HF, QD, HF, 136, false);
  }
  img(IMG_KW1 , kW1, HH, DIM, 72,  false);
  img(IMG_KW2 , kW2, HH, HH,  264, false);
  img(IMG_KW1T, kW1, HH, DIM, 264, true);
  img(IMG_KW2T, kW2, HH, HH,  264, true);
  img(IMG_VW1 , vW1, HH, QD,  40,  false);
  img(IMG_VW2 , vW2, HH, HH,  264, false);
  img(IMG_VW1T, vW1, HH, QD,  264, true);
  img(IMG_VW2T, vW2, HH, HH,  264, true);
}

__global__ void __launch_bounds__(128)
hflow_kernel(const float* __restrict__ hin, const float* __restrict__ u,
  const float* __restrict__ sb1, const float* __restrict__ sb2,
  const float* __restrict__ tb1, const float* __restrict__ tb2,
  const float* __restrict__ kb1, const float* __restrict__ kb2,
  const float* __restrict__ kW3,
  const float* __restrict__ vb1, const float* __restrict__ vb2,
  const float* __restrict__ vW3,
  const float* __restrict__ Ag, const float* __restrict__ Lp,
  const float* __restrict__ Bc,
  float* __restrict__ out)
{
  extern __shared__ __align__(16) char smem[];
  const int tid = threadIdx.x;
  const int wave = tid >> 5, lane = tid & 31;
  const int lh = lane >> 4, lc = lane & 15;

  float*    Sw   = (float*)(smem + SBUF_OFF) + wave * 16 * SLD;
  _Float16* H1   = (_Float16*)(smem + H1_OFF) + wave * 16 * HLD;
  _Float16* H2   = (_Float16*)(smem + H2_OFF) + wave * 16 * HLD;
  _Float16* H3   = (_Float16*)(smem + H3_OFF) + wave * 16 * HLD;
  _Float16* Mlds = (_Float16*)(smem + M_OFF);
  char*     arena = smem + ARENA_OFF;
  float*    cst   = (float*)(smem + CST_OFF);

  float* cb_sb1 = cst;          // 768
  float* cb_sb2 = cst + 768;    // 192
  float* cb_tb1 = cst + 960;    // 768
  float* cb_tb2 = cst + 1728;   // 192
  float* cb_kb1 = cst + 1920;   // 256
  float* cb_kb2 = cst + 2176;   // 256
  float* cb_kw3 = cst + 2432;   // 256
  float* cb_vb1 = cst + 2688;   // 256
  float* cb_vb2 = cst + 2944;   // 256
  float* cb_vw3 = cst + 3200;   // 256
  float* cb_Bc  = cst + 3456;   // 32

  // ---------- stage constants + build M = A - A^T - L L^T ----------
  copy_f32(cb_sb1, sb1, 768, tid);
  copy_f32(cb_sb2, sb2, 192, tid);
  copy_f32(cb_tb1, tb1, 768, tid);
  copy_f32(cb_tb2, tb2, 192, tid);
  copy_f32(cb_kb1, kb1, 256, tid);
  copy_f32(cb_kb2, kb2, 256, tid);
  copy_f32(cb_kw3, kW3, 256, tid);
  copy_f32(cb_vb1, vb1, 256, tid);
  copy_f32(cb_vb2, vb2, 256, tid);
  copy_f32(cb_vw3, vW3, 256, tid);
  copy_f32(cb_Bc,  Bc,  32,  tid);
  {
    float* Lt = (float*)arena;                 // [64][68] f32
    for (int i = tid; i < 64*64; i += 128){
      int r = i >> 6, c0 = i & 63;
      float v = 0.f;
      if (r > c0)       v = Lp[i];
      else if (r == c0) v = softplusf_(Lp[r*64+r]);
      Lt[r*68 + c0] = v;
    }
    __syncthreads();
    for (int i = tid; i < 64*64; i += 128){
      int r = i >> 6, c0 = i & 63;
      float dot = 0.f;
      for (int k = 0; k < 64; ++k) dot += Lt[r*68+k]*Lt[c0*68+k];
      Mlds[r*MLD + c0] = (_Float16)(Ag[r*64+c0] - Ag[c0*64+r] - dot);
    }
    __syncthreads();
  }

  const int rowbase = blockIdx.x * 64 + wave * 16;
  for (int i = lane; i < 16*64; i += 32){
    int m = i >> 6, c0 = i & 63;
    Sw[m*SLD + c0] = hin[(size_t)(rowbase+m)*64 + c0];
  }
  const float uval = u[rowbase + lc];
  memfence();

  // ---------- flow forward ----------
  _Float16* fw_s1 = (_Float16*)(arena);            // [128][40]
  _Float16* fw_s2 = (_Float16*)(arena + 10240);    // [32][136]
  _Float16* fw_t1 = (_Float16*)(arena + 18944);
  _Float16* fw_t2 = (_Float16*)(arena + 29184);
  #pragma unroll 1
  for (int L = 0; L < NLAYERS; ++L){
    __syncthreads();
    stage_img(smem, ARENA_OFF, g_img + IMG_FLOW(L), 18944, 0, tid);
    __syncthreads();
    const int par = L & 1;
    const float* cond = Sw + (par ? QD : 0);
    const int uoff = par ? 0 : QD;
    denseT<0,true ,QD>(fw_s1, 40,  HF, cond, SLD, cb_sb1 + L*HF, H1, HLD, nullptr, nullptr, lane);
    memfence();
    denseT<1,false,HF>(fw_s2, 136, QD, H1,   HLD, cb_sb2 + L*QD, H2, HLD, nullptr, nullptr, lane);
    memfence();
    denseT<0,true ,QD>(fw_t1, 40,  HF, cond, SLD, cb_tb1 + L*HF, H1, HLD, nullptr, nullptr, lane);
    memfence();
    denseT<2,false,HF>(fw_t2, 136, QD, H1,   HLD, cb_tb2 + L*QD, H3, HLD, nullptr, nullptr, lane);
    memfence();
    for (int i = lane; i < 16*QD; i += 32){
      int m = i >> 5, j = i & 31;
      float sv = (float)H2[m*HLD + j], tv = (float)H3[m*HLD + j];
      float xv = Sw[m*SLD + uoff + j];
      Sw[m*SLD + uoff + j] = xv * __expf(sv) + tv;
    }
    memfence();
  }

  // ---------- RK4 over Hamiltonian dynamics ----------
  _Float16* w1f = (_Float16*)(arena);
  _Float16* w2f = (_Float16*)(arena + 36864);
  _Float16* w2t = (_Float16*)(arena);
  _Float16* w1t = (_Float16*)(arena + 135168);

  float z0[32], acc[32];
  #pragma unroll
  for (int t = 0; t < 4; ++t)
    #pragma unroll
    for (int r = 0; r < 8; ++r){
      z0[t*8+r]  = Sw[lc*SLD + 16*t + 8*lh + r];
      acc[t*8+r] = 0.f;
    }

  #pragma unroll 1
  for (int st = 0; st < 4; ++st){
    v8f gh[4];
    #pragma unroll
    for (int t=0;t<4;++t){ v8f z = {}; gh[t] = z; }

    // ===== T net: forward =====
    __syncthreads();
    stage_img(smem, ARENA_OFF,         g_img + IMG_KW1, 18432, 0,   tid);
    stage_img(smem, ARENA_OFF + 36864, g_img + IMG_KW2, 264,   256, tid);
    __syncthreads();
    denseT<3,true ,DIM>(w1f, 72,  HH, Sw, SLD, cb_kb1, H1, HLD, H2, nullptr, lane);
    memfence();
    denseT<4,false,HH >(w2f, HLD, HH, H1, HLD, cb_kb2, H3, HLD, nullptr, cb_kw3, lane);
    memfence();
    // ===== T net: backward =====
    __syncthreads();
    stage_img(smem, ARENA_OFF,          g_img + IMG_KW2T, 264,   256, tid);
    stage_img(smem, ARENA_OFF + 135168, g_img + IMG_KW1T, 16896, 0,   tid);
    __syncthreads();
    grad_hiddenT(H3, H2, w2t, H1, lane);
    memfence();
    #pragma unroll
    for (int t = 0; t < 4; ++t){
      v8f c = gh[t];
      #pragma unroll
      for (int k0 = 0; k0 < HH; k0 += 32)
        c = wmma_f16(a_frag_w(w1t, HLD, 16*t, k0, lane), b_frag_x16(H1, HLD, k0, lane), c);
      gh[t] = c;
    }

    // ===== V net: forward =====
    __syncthreads();
    stage_img(smem, ARENA_OFF,         g_img + IMG_VW1, 10240, 0,   tid);
    stage_img(smem, ARENA_OFF + 36864, g_img + IMG_VW2, 264,   256, tid);
    __syncthreads();
    denseT<3,true ,QD>(w1f, 40,  HH, Sw, SLD, cb_vb1, H1, HLD, H2, nullptr, lane);
    memfence();
    denseT<4,false,HH>(w2f, HLD, HH, H1, HLD, cb_vb2, H3, HLD, nullptr, cb_vw3, lane);
    memfence();
    // ===== V net: backward =====
    __syncthreads();
    stage_img(smem, ARENA_OFF,          g_img + IMG_VW2T, 264,  256, tid);
    stage_img(smem, ARENA_OFF + 135168, g_img + IMG_VW1T, 8448, 0,   tid);
    __syncthreads();
    grad_hiddenT(H3, H2, w2t, H1, lane);
    memfence();
    #pragma unroll
    for (int t = 0; t < 2; ++t){
      v8f c = gh[t];
      #pragma unroll
      for (int k0 = 0; k0 < HH; k0 += 32)
        c = wmma_f16(a_frag_w(w1t, HLD, 16*t, k0, lane), b_frag_x16(H1, HLD, k0, lane), c);
      gh[t] = c;
    }

    // ===== dz^T = M @ gH^T (+ Bu on p), RK4 update =====
    _Float16* GHh = H1;                     // reuse as f16 [16][GLD]
    #pragma unroll
    for (int t=0;t<4;++t){
      v8h g;
      #pragma unroll
      for (int r=0;r<8;++r) g[r] = (_Float16)gh[t][r];
      *(v8h*)(GHh + lc*GLD + 16*t + 8*lh) = g;
    }
    memfence();
    const float win = (st==0||st==1) ? 0.5f*DTS : (st==2 ? DTS : 0.f);
    const float wa  = (st==1||st==2) ? 2.f : 1.f;
    #pragma unroll
    for (int t=0;t<4;++t){
      v8f c = {};
      #pragma unroll
      for (int k0=0;k0<DIM;k0+=32)
        c = wmma_f16(a_frag_w(Mlds, MLD, 16*t, k0, lane), b_frag_x16(GHh, GLD, k0, lane), c);
      #pragma unroll
      for (int r=0;r<8;++r){
        float d = c[r];
        if (t >= 2) d += uval * cb_Bc[16*(t-2) + 8*lh + r];
        acc[t*8+r] += wa * d;
        if (st < 3) Sw[lc*SLD + 16*t + 8*lh + r] = z0[t*8+r] + win * d;
      }
    }
    memfence();
  }
  #pragma unroll
  for (int t=0;t<4;++t)
    #pragma unroll
    for (int r=0;r<8;++r)
      Sw[lc*SLD + 16*t + 8*lh + r] = z0[t*8+r] + (DTS/6.f)*acc[t*8+r];
  memfence();

  // ---------- flow inverse ----------
  #pragma unroll 1
  for (int L = NLAYERS-1; L >= 0; --L){
    __syncthreads();
    stage_img(smem, ARENA_OFF, g_img + IMG_FLOW(L), 18944, 0, tid);
    __syncthreads();
    const int par = L & 1;
    const float* cond = Sw + (par ? QD : 0);
    const int uoff = par ? 0 : QD;
    denseT<0,true ,QD>(fw_s1, 40,  HF, cond, SLD, cb_sb1 + L*HF, H1, HLD, nullptr, nullptr, lane);
    memfence();
    denseT<1,false,HF>(fw_s2, 136, QD, H1,   HLD, cb_sb2 + L*QD, H2, HLD, nullptr, nullptr, lane);
    memfence();
    denseT<0,true ,QD>(fw_t1, 40,  HF, cond, SLD, cb_tb1 + L*HF, H1, HLD, nullptr, nullptr, lane);
    memfence();
    denseT<2,false,HF>(fw_t2, 136, QD, H1,   HLD, cb_tb2 + L*QD, H3, HLD, nullptr, nullptr, lane);
    memfence();
    for (int i = lane; i < 16*QD; i += 32){
      int m = i >> 5, j = i & 31;
      float sv = (float)H2[m*HLD + j], tv = (float)H3[m*HLD + j];
      float yv = Sw[m*SLD + uoff + j];
      Sw[m*SLD + uoff + j] = (yv - tv) * __expf(-sv);
    }
    memfence();
  }

  for (int i = lane; i < 16*64; i += 32){
    int m = i >> 6, c0 = i & 63;
    out[(size_t)(rowbase+m)*64 + c0] = Sw[m*SLD + c0];
  }
}

extern "C" void kernel_launch(void* const* d_in, const int* in_sizes, int n_in,
                              void* d_out, int out_size, void* d_ws, size_t ws_size,
                              hipStream_t stream) {
  (void)in_sizes; (void)n_in; (void)d_ws; (void)ws_size; (void)out_size;
  const float* h    = (const float*)d_in[0];
  const float* u    = (const float*)d_in[1];
  const float* sW1  = (const float*)d_in[2];
  const float* sb1  = (const float*)d_in[3];
  const float* sW2  = (const float*)d_in[4];
  const float* sb2  = (const float*)d_in[5];
  const float* tW1  = (const float*)d_in[6];
  const float* tb1  = (const float*)d_in[7];
  const float* tW2  = (const float*)d_in[8];
  const float* tb2  = (const float*)d_in[9];
  const float* kW1  = (const float*)d_in[10];
  const float* kb1  = (const float*)d_in[11];
  const float* kW2  = (const float*)d_in[12];
  const float* kb2  = (const float*)d_in[13];
  const float* kW3  = (const float*)d_in[14];
  const float* vW1  = (const float*)d_in[16];
  const float* vb1  = (const float*)d_in[17];
  const float* vW2  = (const float*)d_in[18];
  const float* vb2  = (const float*)d_in[19];
  const float* vW3  = (const float*)d_in[20];
  const float* A    = (const float*)d_in[22];
  const float* Lp   = (const float*)d_in[23];
  const float* Bc   = (const float*)d_in[24];
  float* out = (float*)d_out;

  prep_kernel<<<dim3(128), dim3(256), 0, stream>>>(sW1, sW2, tW1, tW2, kW1, kW2, vW1, vW2);

  dim3 grid(BATCH / 64), block(128);
  hflow_kernel<<<grid, block, SMEM_BYTES, stream>>>(
      h, u, sb1, sb2, tb1, tb2, kb1, kb2, kW3, vb1, vb2, vW3,
      A, Lp, Bc, out);
}
